// KeyNodeAttention_70377334112384
// MI455X (gfx1250) — compile-verified, matrix operand order checked by reading, not run on hardware
//
#include <hip/hip_runtime.h>
#include <hip/hip_bf16.h>

// ---- problem constants (match reference) ----
#define B_  8
#define T_  12
#define F_  96          // B_*T_
#define NN  4000
#define D_  64
#define TKK 64
#define H_  8
#define EPSF 1e-5f

typedef __attribute__((ext_vector_type(16))) _Float16 v16h;
typedef __attribute__((ext_vector_type(8)))  float    v8f;

__device__ __forceinline__ v8f vzero8() {
    v8f z = {0.f,0.f,0.f,0.f,0.f,0.f,0.f,0.f};
    return z;
}

// K offset pattern for 16-bit A (16x32) tiles, per ISA 7.12.2.
__device__ __forceinline__ int k_off(int v, int half) {
    return (v < 4) ? (2*v + 8*half) : (16 + 2*(v-4) + 8*half);
}

// Load a 16x32 f32 tile (rows = lane%16, cols kbase..kbase+31) converting to
// f16 in WMMA A layout. Also used for B tiles: for y = x * W^T with W stored
// (o,c) row-major, B[k][o] = W[o][k], so loading W rows in "A layout" with
// lane%16 = o gives exactly the B operand.
__device__ __forceinline__ v16h load_ab_f32(const float* __restrict__ p0,
                                            int stride, int kbase, int lane) {
    const int mr = lane & 15, half = lane >> 4;
    const float* p = p0 + mr * stride + kbase;
    v16h t;
#pragma unroll
    for (int v = 0; v < 8; ++v) {
        const int k0 = k_off(v, half);
        t[2*v]   = (_Float16)p[k0];
        t[2*v+1] = (_Float16)p[k0+1];
    }
    return t;
}

__device__ __forceinline__ v8f wmma_f16(v16h a, v16h b, v8f c) {
    return __builtin_amdgcn_wmma_f32_16x16x32_f16(false, a, false, b,
                                                  (short)0, c, false, false);
}

// ------------------------------------------------------------------
// Kernel 1: Xc=[X|STE] -> FC1(bn,relu) -> FC2(bn) = Xw  (stored to ws)
//           -> importance MLP -> per-frame scores
// One wave (32 threads) per (frame, 16-node tile).
// ------------------------------------------------------------------
__global__ __launch_bounds__(32)
void frontend_kernel(const float* __restrict__ X, const float* __restrict__ STE,
                     const float* __restrict__ W1w, const float* __restrict__ W1b,
                     const float* __restrict__ W1bn,
                     const float* __restrict__ W2w, const float* __restrict__ W2b,
                     const float* __restrict__ W2bn,
                     const float* __restrict__ iw1, const float* __restrict__ ib1,
                     const float* __restrict__ iw2, const float* __restrict__ ib2,
                     const float* __restrict__ iw3, const float* __restrict__ ib3,
                     float* __restrict__ XwG, float* __restrict__ scoresG)
{
    __shared__ __align__(16) float lt[16*64];
    const int lane  = threadIdx.x;
    const int f     = blockIdx.y;
    const int node0 = blockIdx.x * 16;
    const int n16 = lane & 15, half = lane >> 4;
    const size_t rowbase = ((size_t)f * NN + node0) * D_;
    const float* Xb = X  + rowbase;
    const float* Sb = STE + rowbase;

    // ---- FC1: (16x128) x (128x64) ----
    v8f acc[4];
#pragma unroll
    for (int ot = 0; ot < 4; ++ot) acc[ot] = vzero8();
#pragma unroll
    for (int ks = 0; ks < 4; ++ks) {
        const float* src = (ks < 2) ? Xb : Sb;
        v16h a = load_ab_f32(src, D_, (ks & 1) * 32, lane);
#pragma unroll
        for (int ot = 0; ot < 4; ++ot) {
            v16h b = load_ab_f32(W1w + ot*16*128, 128, ks*32, lane);
            acc[ot] = wmma_f16(a, b, acc[ot]);
        }
    }
    // bias + BN + ReLU, transpose into LDS [m][o]
#pragma unroll
    for (int ot = 0; ot < 4; ++ot) {
        const int o = ot*16 + n16;
        const float bia = W1b[o];
        const float sc  = rsqrtf(W1bn[192+o] + EPSF) * W1bn[o];
        const float mu  = W1bn[128+o], be = W1bn[64+o];
#pragma unroll
        for (int r = 0; r < 8; ++r) {
            float y = (acc[ot][r] + bia - mu) * sc + be;
            lt[(r + 8*half)*64 + o] = fmaxf(y, 0.f);
        }
    }
    __syncthreads();

    // ---- FC2: (16x64) x (64x64) ----
#pragma unroll
    for (int ot = 0; ot < 4; ++ot) acc[ot] = vzero8();
#pragma unroll
    for (int ks = 0; ks < 2; ++ks) {
        v16h a = load_ab_f32(lt, 64, ks*32, lane);
#pragma unroll
        for (int ot = 0; ot < 4; ++ot) {
            v16h b = load_ab_f32(W2w + ot*16*64, 64, ks*32, lane);
            acc[ot] = wmma_f16(a, b, acc[ot]);
        }
    }
    __syncthreads();
#pragma unroll
    for (int ot = 0; ot < 4; ++ot) {
        const int o = ot*16 + n16;
        const float bia = W2b[o];
        const float sc  = rsqrtf(W2bn[192+o] + EPSF) * W2bn[o];
        const float mu  = W2bn[128+o], be = W2bn[64+o];
#pragma unroll
        for (int r = 0; r < 8; ++r)
            lt[(r + 8*half)*64 + o] = (acc[ot][r] + bia - mu) * sc + be;
    }
    __syncthreads();

    // coalesced store of Xw tile to workspace
#pragma unroll
    for (int i = 0; i < 8; ++i) {
        float4 v4 = *(const float4*)&lt[i*128 + lane*4];
        *(float4*)(XwG + rowbase + i*128 + lane*4) = v4;
    }

    // ---- importance MLP layer 1: 64->64, ReLU ----
#pragma unroll
    for (int ot = 0; ot < 4; ++ot) acc[ot] = vzero8();
#pragma unroll
    for (int ks = 0; ks < 2; ++ks) {
        v16h a = load_ab_f32(lt, 64, ks*32, lane);
#pragma unroll
        for (int ot = 0; ot < 4; ++ot) {
            v16h b = load_ab_f32(iw1 + ot*16*64, 64, ks*32, lane);
            acc[ot] = wmma_f16(a, b, acc[ot]);
        }
    }
    __syncthreads();
#pragma unroll
    for (int ot = 0; ot < 4; ++ot) {
        const int o = ot*16 + n16;
        const float bia = ib1[o];
#pragma unroll
        for (int r = 0; r < 8; ++r)
            lt[(r + 8*half)*64 + o] = fmaxf(acc[ot][r] + bia, 0.f);
    }
    __syncthreads();

    // ---- layer 2: 64->32, ReLU ----
    v8f acc2[2];
    acc2[0] = vzero8(); acc2[1] = vzero8();
#pragma unroll
    for (int ks = 0; ks < 2; ++ks) {
        v16h a = load_ab_f32(lt, 64, ks*32, lane);
#pragma unroll
        for (int ot = 0; ot < 2; ++ot) {
            v16h b = load_ab_f32(iw2 + ot*16*64, 64, ks*32, lane);
            acc2[ot] = wmma_f16(a, b, acc2[ot]);
        }
    }
    __syncthreads();
#pragma unroll
    for (int ot = 0; ot < 2; ++ot) {
        const int o = ot*16 + n16;
        const float bia = ib2[o];
#pragma unroll
        for (int r = 0; r < 8; ++r)
            lt[(r + 8*half)*64 + o] = fmaxf(acc2[ot][r] + bia, 0.f);
    }
    __syncthreads();

    // ---- layer 3: 32->1 dot, per-frame score ----
    if (lane < 16) {
        float s = ib3[0];
#pragma unroll 4
        for (int o = 0; o < 32; ++o) s += lt[lane*64 + o] * iw3[o];
        scoresG[(size_t)f * NN + node0 + lane] = s;
    }
}

// ------------------------------------------------------------------
// Kernel 2: reduce scores over frames, top-64 (stable tie-break),
//           sort ascending. Single block.
// ------------------------------------------------------------------
__global__ __launch_bounds__(256)
void topk_kernel(const float* __restrict__ scoresG, int* __restrict__ idxG)
{
    __shared__ float imp[NN];
    __shared__ float bv[256];
    __shared__ int   bi[256];
    __shared__ int   sel[TKK];
    const int tid = threadIdx.x;

    for (int n = tid; n < NN; n += 256) {
        float s = 0.f;
        for (int f = 0; f < F_; ++f) s += scoresG[(size_t)f * NN + n];
        imp[n] = s;
    }
    __syncthreads();

    for (int it = 0; it < TKK; ++it) {
        float best = -1e30f; int bidx = 0x7fffffff;
        for (int n = tid; n < NN; n += 256) {
            float v = imp[n];
            if (v > best || (v == best && n < bidx)) { best = v; bidx = n; }
        }
        bv[tid] = best; bi[tid] = bidx;
        __syncthreads();
        for (int s = 128; s > 0; s >>= 1) {
            if (tid < s) {
                float v2 = bv[tid+s]; int i2 = bi[tid+s];
                if (v2 > bv[tid] || (v2 == bv[tid] && i2 < bi[tid])) {
                    bv[tid] = v2; bi[tid] = i2;
                }
            }
            __syncthreads();
        }
        if (tid == 0) { sel[it] = bi[0]; imp[bi[0]] = -1e30f; }
        __syncthreads();
    }

    if (tid == 0) {
        for (int i = 1; i < TKK; ++i) {        // insertion sort ascending
            int key = sel[i], j = i - 1;
            while (j >= 0 && sel[j] > key) { sel[j+1] = sel[j]; --j; }
            sel[j+1] = key;
        }
        for (int i = 0; i < TKK; ++i) idxG[i] = sel[i];
    }
}

// ------------------------------------------------------------------
// Kernel 3: per frame: gather keyX, QKV (WMMA, bias+BN), MHA softmax.
// One block (256 thr = 8 waves) per frame.
// ------------------------------------------------------------------
__global__ __launch_bounds__(256)
void attn_kernel(const float* __restrict__ XwG, const int* __restrict__ idxG,
                 const float* __restrict__ qkvw, const float* __restrict__ qkvb,
                 const float* __restrict__ qkvbn, float* __restrict__ attOut)
{
    __shared__ __align__(16) float kx[64*64];
    __shared__ float qs[64*64];
    __shared__ float kk[64*64];
    __shared__ float vv[64*64];
    const int tid = threadIdx.x;
    const int f = blockIdx.x;

    for (int i = tid; i < 64*64; i += 256) {
        const int row = i >> 6, col = i & 63;
        const int node = idxG[row];
        kx[i] = XwG[((size_t)f * NN + node) * D_ + col];
    }
    __syncthreads();

    const int w = tid >> 5, lane = tid & 31;
    const int n16 = lane & 15, half = lane >> 4;

    // 12 wave-jobs: {Q,K,V} x 4 row-tiles of 16
    for (int job = w; job < 12; job += 8) {
        const int mat = job >> 2, mt = job & 3;
        v8f acc[4];
#pragma unroll
        for (int ot = 0; ot < 4; ++ot) acc[ot] = vzero8();
#pragma unroll
        for (int ks = 0; ks < 2; ++ks) {
            v16h a = load_ab_f32(kx + mt*16*64, 64, ks*32, lane);
#pragma unroll
            for (int ot = 0; ot < 4; ++ot) {
                v16h b = load_ab_f32(qkvw + mat*4096 + ot*16*64, 64, ks*32, lane);
                acc[ot] = wmma_f16(a, b, acc[ot]);
            }
        }
        float* dst = (mat == 0) ? qs : (mat == 1) ? kk : vv;
        const float* bn = qkvbn + mat*256;
#pragma unroll
        for (int ot = 0; ot < 4; ++ot) {
            const int o = ot*16 + n16;
            const float bia = qkvb[mat*64 + o];
            const float sc  = rsqrtf(bn[192+o] + EPSF) * bn[o];
            const float mu  = bn[128+o], be = bn[64+o];
#pragma unroll
            for (int r = 0; r < 8; ++r)
                dst[(mt*16 + r + 8*half)*64 + o] = (acc[ot][r] + bia - mu) * sc + be;
        }
    }
    __syncthreads();

    // one wave per head; each lane handles q = lane and q = lane+32
    const int h = tid >> 5;
    const float rs = 0.35355339059327373f;   // 1/sqrt(8)
    for (int q = lane; q < 64; q += 32) {
        float qv[8];
#pragma unroll
        for (int d = 0; d < 8; ++d) qv[d] = qs[q*64 + h*8 + d];
        float mx = -1e30f;
        for (int k = 0; k < 64; ++k) {
            float dv = 0.f;
#pragma unroll
            for (int d = 0; d < 8; ++d) dv += qv[d] * kk[k*64 + h*8 + d];
            mx = fmaxf(mx, dv * rs);
        }
        float den = 0.f;
        float ov[8];
#pragma unroll
        for (int d = 0; d < 8; ++d) ov[d] = 0.f;
        for (int k = 0; k < 64; ++k) {
            float dv = 0.f;
#pragma unroll
            for (int d = 0; d < 8; ++d) dv += qv[d] * kk[k*64 + h*8 + d];
            const float e = __expf(dv * rs - mx);
            den += e;
#pragma unroll
            for (int d = 0; d < 8; ++d) ov[d] += e * vv[k*64 + h*8 + d];
        }
        const float inv = 1.f / den;
#pragma unroll
        for (int d = 0; d < 8; ++d)
            attOut[((size_t)f*TKK + q)*D_ + h*8 + d] = ov[d] * inv;
    }
}

// ------------------------------------------------------------------
// Kernel 4: fused = FC([Xw | scatter(attOut)], fus_w) + bias + BN + ReLU
// One wave per (frame, 16-node tile); binary search over sorted idx.
// ------------------------------------------------------------------
__device__ __forceinline__ int bsearch64(const int* s, int key) {
    int lo = 0, hi = 63;
    while (lo <= hi) {
        int mid = (lo + hi) >> 1;
        int v = s[mid];
        if (v == key) return mid;
        if (v < key) lo = mid + 1; else hi = mid - 1;
    }
    return -1;
}

__global__ __launch_bounds__(32)
void fuse_kernel(const float* __restrict__ XwG, const float* __restrict__ attOut,
                 const int* __restrict__ idxG,
                 const float* __restrict__ fw, const float* __restrict__ fb,
                 const float* __restrict__ fbn, float* __restrict__ out)
{
    __shared__ __align__(16) float lt[16*64];
    __shared__ int sidx[64];
    const int lane = threadIdx.x;
    sidx[lane]      = idxG[lane];
    sidx[lane + 32] = idxG[lane + 32];
    __syncthreads();

    const int f = blockIdx.y;
    const int node0 = blockIdx.x * 16;
    const int n16 = lane & 15, half = lane >> 4;
    const size_t rowbase = ((size_t)f * NN + node0) * D_;

    const int j = bsearch64(sidx, node0 + n16);
    const float* fullrow = (j >= 0) ? attOut + ((size_t)f*TKK + j)*D_ : nullptr;

    v8f acc[4];
#pragma unroll
    for (int ot = 0; ot < 4; ++ot) acc[ot] = vzero8();
#pragma unroll
    for (int ks = 0; ks < 4; ++ks) {
        v16h a;
        if (ks < 2) {
            a = load_ab_f32(XwG + rowbase, D_, ks*32, lane);
        } else {
            const int kb = (ks - 2) * 32;
#pragma unroll
            for (int v = 0; v < 8; ++v) {
                const int k0 = k_off(v, half);
                a[2*v]   = fullrow ? (_Float16)fullrow[kb + k0]     : (_Float16)0.f;
                a[2*v+1] = fullrow ? (_Float16)fullrow[kb + k0 + 1] : (_Float16)0.f;
            }
        }
#pragma unroll
        for (int ot = 0; ot < 4; ++ot) {
            v16h b = load_ab_f32(fw + ot*16*128, 128, ks*32, lane);
            acc[ot] = wmma_f16(a, b, acc[ot]);
        }
    }

#pragma unroll
    for (int ot = 0; ot < 4; ++ot) {
        const int o = ot*16 + n16;
        const float bia = fb[o];
        const float sc  = rsqrtf(fbn[192+o] + EPSF) * fbn[o];
        const float mu  = fbn[128+o], be = fbn[64+o];
#pragma unroll
        for (int r = 0; r < 8; ++r) {
            float y = (acc[ot][r] + bia - mu) * sc + be;
            lt[(r + 8*half)*64 + o] = fmaxf(y, 0.f);
        }
    }
    __syncthreads();

#pragma unroll
    for (int i = 0; i < 8; ++i) {
        float4 v4 = *(const float4*)&lt[i*128 + lane*4];
        *(float4*)(out + rowbase + i*128 + lane*4) = v4;
    }
}

// ------------------------------------------------------------------
extern "C" void kernel_launch(void* const* d_in, const int* in_sizes, int n_in,
                              void* d_out, int out_size, void* d_ws, size_t ws_size,
                              hipStream_t stream)
{
    (void)in_sizes; (void)n_in; (void)out_size; (void)ws_size;
    const float* X    = (const float*)d_in[0];
    /* d_in[1] = adj_matrix, unused */
    const float* STE  = (const float*)d_in[2];
    const float* iw1  = (const float*)d_in[3];
    const float* ib1  = (const float*)d_in[4];
    const float* iw2  = (const float*)d_in[5];
    const float* ib2  = (const float*)d_in[6];
    const float* iw3  = (const float*)d_in[7];
    const float* ib3  = (const float*)d_in[8];
    const float* W1w  = (const float*)d_in[9];
    const float* W1b  = (const float*)d_in[10];
    const float* W1bn = (const float*)d_in[11];
    const float* W2w  = (const float*)d_in[12];
    const float* W2b  = (const float*)d_in[13];
    const float* W2bn = (const float*)d_in[14];
    const float* qkvw = (const float*)d_in[15];
    const float* qkvb = (const float*)d_in[16];
    const float* qkvbn= (const float*)d_in[17];
    const float* fw   = (const float*)d_in[18];
    const float* fb   = (const float*)d_in[19];
    const float* fbn  = (const float*)d_in[20];
    float* out = (float*)d_out;

    float* XwG     = (float*)d_ws;
    float* scoresG = XwG + (size_t)F_ * NN * D_;
    float* attOut  = scoresG + (size_t)F_ * NN;
    int*   idxG    = (int*)(attOut + (size_t)F_ * TKK * D_);

    dim3 gtiles(NN / 16, F_);
    frontend_kernel<<<gtiles, 32, 0, stream>>>(X, STE, W1w, W1b, W1bn,
                                               W2w, W2b, W2bn,
                                               iw1, ib1, iw2, ib2, iw3, ib3,
                                               XwG, scoresG);
    topk_kernel<<<1, 256, 0, stream>>>(scoresG, idxG);
    attn_kernel<<<F_, 256, 0, stream>>>(XwG, idxG, qkvw, qkvb, qkvbn, attOut);
    fuse_kernel<<<gtiles, 32, 0, stream>>>(XwG, attOut, idxG, fw, fb, fbn, out);
}
